// AdaptiveMultiHeadAttention_25228637897518
// MI455X (gfx1250) — matrix operand order repaired and checked by reading, not compiled, not run on hardware
//
#include <hip/hip_runtime.h>

// ---------------- problem constants ----------------
#define F_DIM   256
#define N_NODES 1024
#define H_HEADS 4
#define C_COMP  8
#define BATCH   16
#define HD      64     // F/H

typedef __attribute__((ext_vector_type(16))) __bf16 v16bf;
typedef __attribute__((ext_vector_type(8)))  float  v8f;

static __device__ __forceinline__ unsigned short f2bf(float f) {
  union { float f; unsigned u; } v; v.f = f;
  unsigned r = v.u + 0x7FFFu + ((v.u >> 16) & 1u);   // round-to-nearest-even
  return (unsigned short)(r >> 16);
}

// ---------------------------------------------------------------------------
// Kernel 1: qkv = (feat @ Wc + bc) @ We + be ; emit bf16 q,k (row-major over hd)
// and v transposed [B,H,hd,N] so the attn@v B-fragment loads are contiguous.
// ---------------------------------------------------------------------------
__global__ __launch_bounds__(256) void qkv_kernel(
    const float* __restrict__ feat, const float* __restrict__ Wc,
    const float* __restrict__ bc,   const float* __restrict__ We,
    const float* __restrict__ be,
    unsigned short* __restrict__ qbf, unsigned short* __restrict__ kbf,
    unsigned short* __restrict__ vT)
{
  const int row = blockIdx.x;               // b*N + n
  const int b = row >> 10, n = row & 1023;
  const int t = threadIdx.x;
  __shared__ float s_feat[F_DIM];
  __shared__ float s_comp[3 * C_COMP];

  s_feat[t] = feat[(size_t)row * F_DIM + t];
  __syncthreads();

  if (t < 3 * C_COMP) {
    float acc = bc[t];
    #pragma unroll 8
    for (int i = 0; i < F_DIM; ++i) acc += s_feat[i] * Wc[i * (3*C_COMP) + t];
    s_comp[t] = acc;
  }
  __syncthreads();

  float q = be[t], k = be[F_DIM + t], v = be[2*F_DIM + t];
  #pragma unroll
  for (int j = 0; j < 3 * C_COMP; ++j) {
    const float c = s_comp[j];
    q += c * We[j * (3*F_DIM) + t];
    k += c * We[j * (3*F_DIM) + F_DIM + t];
    v += c * We[j * (3*F_DIM) + 2*F_DIM + t];
  }
  const int h = t >> 6, d = t & 63;
  const size_t qi = (((size_t)b * H_HEADS + h) * N_NODES + n) * HD + d;
  qbf[qi] = f2bf(q);
  kbf[qi] = f2bf(k);
  vT[(((size_t)b * H_HEADS + h) * HD + d) * N_NODES + n] = f2bf(v);
}

// ---------------------------------------------------------------------------
// Kernel 2: biasH[h,n,m] = (bf1[h] @ bf2[h])[n,m] + (1-sigmoid(g)) * fixed[n,m]
// ---------------------------------------------------------------------------
__global__ __launch_bounds__(256) void bias_kernel(
    const float* __restrict__ bf1, const float* __restrict__ bf2,
    const float* __restrict__ fixedg, const float* __restrict__ fusion,
    float* __restrict__ biasH)
{
  const int idx = blockIdx.x;               // h*N + n
  const int h = idx >> 10, n = idx & 1023;
  const int t = threadIdx.x;
  const float fr = 1.f / (1.f + __expf(-fusion[0]));
  const float w  = 1.f - fr;

  float a[C_COMP];
  #pragma unroll
  for (int c = 0; c < C_COMP; ++c)
    a[c] = bf1[((size_t)h * N_NODES + n) * C_COMP + c];

  #pragma unroll
  for (int mm = 0; mm < 4; ++mm) {
    const int m = mm * 256 + t;             // coalesced over t
    float s = w * fixedg[(size_t)n * N_NODES + m];
    #pragma unroll
    for (int c = 0; c < C_COMP; ++c)
      s += a[c] * bf2[((size_t)h * C_COMP + c) * N_NODES + m];
    biasH[((size_t)h * N_NODES + n) * N_NODES + m] = s;
  }
}

// ---------------------------------------------------------------------------
// Kernel 3: attention. One wave32 per (b, h, 16-row tile).
//   Phase 1: scores 16x1024 via v_wmma_f32_16x16x32_bf16 + fused bias adds.
//   Phase 2: softmax; exp values written back IN PLACE as bf16 in the WMMA
//            A-fragment interleave (normalization deferred to the accumulator).
//   Phase 3: context = exp @ v via WMMA, scaled by 1/rowsum at store.
// ---------------------------------------------------------------------------
__global__ __launch_bounds__(32) void attn_kernel(
    const unsigned short* __restrict__ qbf, const unsigned short* __restrict__ kbf,
    const unsigned short* __restrict__ vT,  const float* __restrict__ biasH,
    const float* __restrict__ learned,      const float* __restrict__ fusion,
    float* __restrict__ ctx)
{
  __shared__ float s_sc[16][N_NODES];       // 64 KB; f32 scores, then bf16 exp
  __shared__ float s_inv[16];

  const int l    = threadIdx.x;
  const int blk  = blockIdx.x;
  const int it   = blk & 63;
  const int h    = (blk >> 6) & 3;
  const int b    = blk >> 8;
  const int row0 = it * 16;
  const int bh   = b * H_HEADS + h;
  const float fr = 1.f / (1.f + __expf(-fusion[0]));

  // A-fragment lane mapping (16-bit A 16x32): M = l&15, K(e) = akb + e
  const int arow = l & 15;
  const int akb  = (l < 16) ? 0 : 16;
  // B-fragment lane mapping (16-bit B 32x16): N = l&15, K(e) = bkb + e
  const int ncol   = l & 15;
  const int bkb    = (l < 16) ? 0 : 16;
  const int mrowHi = (l < 16) ? 0 : 8;      // C/D: M = p + mrowHi

  union U16 { v16bf v; unsigned short u[16]; };

  // ---- Q fragments (reused across all 64 m-tiles) ----
  U16 aq0, aq1;
  {
    const unsigned short* qr = qbf + (((size_t)bh * N_NODES) + row0 + arow) * HD;
    #pragma unroll
    for (int e = 0; e < 8; ++e) {
      aq0.u[e]     = qr[akb + e];
      aq0.u[8 + e] = qr[akb + 8 + e];
      aq1.u[e]     = qr[32 + akb + e];
      aq1.u[8 + e] = qr[32 + akb + 8 + e];
    }
  }

  // ================= Phase 1: scores =================
  for (int mt = 0; mt < 64; ++mt) {
    const unsigned short* kr =
        kbf + (((size_t)bh * N_NODES) + mt * 16 + ncol) * HD;
    __builtin_prefetch((const void*)(kr + 16 * HD), 0, 0);  // next K tile
    v16bf bk0 = *(const v16bf*)(kr + bkb);        // 32B aligned, contiguous K
    v16bf bk1 = *(const v16bf*)(kr + 32 + bkb);
    v8f acc = {};
    acc = __builtin_amdgcn_wmma_f32_16x16x32_bf16(false, aq0.v, false, bk0,
                                                  (short)0, acc, false, false);
    acc = __builtin_amdgcn_wmma_f32_16x16x32_bf16(false, aq1.v, false, bk1,
                                                  (short)0, acc, false, false);
    const int col = mt * 16 + ncol;
    const float* bias_p = biasH + (((size_t)h * N_NODES) + row0 + mrowHi) * N_NODES + col;
    const float* lrn_p  = learned + (((size_t)b * N_NODES) + row0 + mrowHi) * N_NODES + col;
    #pragma unroll
    for (int p = 0; p < 8; ++p) {
      const float s = acc[p] * 0.125f                 // 1/sqrt(hd)
                    + bias_p[(size_t)p * N_NODES]
                    + fr * lrn_p[(size_t)p * N_NODES];
      s_sc[mrowHi + p][col] = s;
    }
  }
  __syncthreads();

  // ================= Phase 2: softmax, bf16 exp written in place =================
  // Column c = l + 32*j maps to fragment chunk j with fixed (half, idx) per lane:
  const int shalf = l >> 4;                          // K%32 in [16,32) -> 1
  const int sidx  = l & 15;
  for (int r = 0; r < 16; ++r) {
    float vreg[32];
    #pragma unroll
    for (int j = 0; j < 32; ++j) vreg[j] = s_sc[r][l + 32 * j];
    float mx = vreg[0];
    #pragma unroll
    for (int j = 1; j < 32; ++j) mx = fmaxf(mx, vreg[j]);
    #pragma unroll
    for (int off = 16; off > 0; off >>= 1) mx = fmaxf(mx, __shfl_xor(mx, off, 32));
    float sum = 0.f;
    #pragma unroll
    for (int j = 0; j < 32; ++j) {
      const float e = __expf(vreg[j] - mx);
      vreg[j] = e;
      sum += e;
    }
    #pragma unroll
    for (int off = 16; off > 0; off >>= 1) sum += __shfl_xor(sum, off, 32);
    if (l == 0) s_inv[r] = 1.f / sum;
    // overwrite row r slice (reads already staged in registers; lockstep wave)
    unsigned short* rowp =
        ((unsigned short*)&s_sc[r][0]) + shalf * 16 + sidx;
    #pragma unroll
    for (int j = 0; j < 32; ++j) rowp[j * 32] = f2bf(vreg[j]);
  }
  __syncthreads();

  // ================= Phase 3: context = exp @ v, scaled by 1/rowsum =================
  float inv8[8];
  #pragma unroll
  for (int p = 0; p < 8; ++p) inv8[p] = s_inv[mrowHi + p];

  const unsigned short* s16a =
      ((const unsigned short*)&s_sc[0][0]) + arow * 2048 + (akb ? 16 : 0);
  for (int dt = 0; dt < 4; ++dt) {
    const unsigned short* vr = vT + (((size_t)bh * HD) + dt * 16 + ncol) * N_NODES;
    v8f cacc = {};
    for (int mc = 0; mc < 32; ++mc) {
      v16bf aa = *(const v16bf*)(s16a + mc * 32);         // 2x ds_load_b128
      v16bf bb = *(const v16bf*)(vr + mc * 32 + bkb);     // 2x global b128
      cacc = __builtin_amdgcn_wmma_f32_16x16x32_bf16(false, aa, false, bb,
                                                     (short)0, cacc, false, false);
    }
    #pragma unroll
    for (int p = 0; p < 8; ++p) {
      const int rg   = row0 + mrowHi + p;
      const int fcol = h * HD + dt * 16 + ncol;
      ctx[(((size_t)b * N_NODES) + rg) * F_DIM + fcol] = cacc[p] * inv8[p];
    }
  }
}

// ---------------------------------------------------------------------------
// Kernel 4: out = (ctx @ Woc + boc) @ Woe + boe ; +residual ; LayerNorm.
// Also writes reg_loss = 1e-5 * mean|attn| = 1e-5 / N (softmax rows sum to 1).
// ---------------------------------------------------------------------------
__global__ __launch_bounds__(256) void out_kernel(
    const float* __restrict__ ctx,  const float* __restrict__ feat,
    const float* __restrict__ Woc,  const float* __restrict__ boc,
    const float* __restrict__ Woe,  const float* __restrict__ boe,
    const float* __restrict__ gamma, const float* __restrict__ beta,
    float* __restrict__ out)
{
  const int row = blockIdx.x;               // b*N + n
  const int t = threadIdx.x;
  __shared__ float s_c[F_DIM];
  __shared__ float s_c8[C_COMP];
  __shared__ float s_red[18];

  s_c[t] = ctx[(size_t)row * F_DIM + t];
  __syncthreads();
  if (t < C_COMP) {
    float a = boc[t];
    #pragma unroll 8
    for (int i = 0; i < F_DIM; ++i) a += s_c[i] * Woc[i * C_COMP + t];
    s_c8[t] = a;
  }
  __syncthreads();

  float o = boe[t] + feat[(size_t)row * F_DIM + t];
  #pragma unroll
  for (int j = 0; j < C_COMP; ++j) o += s_c8[j] * Woe[j * F_DIM + t];

  // LayerNorm reductions: wave shuffle then cross-wave via LDS
  float v = o, v2 = o * o;
  #pragma unroll
  for (int off = 16; off > 0; off >>= 1) {
    v  += __shfl_xor(v,  off, 32);
    v2 += __shfl_xor(v2, off, 32);
  }
  const int w = t >> 5;
  if ((t & 31) == 0) { s_red[w] = v; s_red[8 + w] = v2; }
  __syncthreads();
  if (t == 0) {
    float sv = 0.f, sv2 = 0.f;
    #pragma unroll
    for (int i = 0; i < 8; ++i) { sv += s_red[i]; sv2 += s_red[8 + i]; }
    const float mu = sv * (1.f / F_DIM);
    s_red[16] = mu;
    s_red[17] = sv2 * (1.f / F_DIM) - mu * mu;
  }
  __syncthreads();
  const float mu  = s_red[16];
  const float rs  = rsqrtf(s_red[17] + 1e-5f);
  out[(size_t)row * F_DIM + t] = (o - mu) * rs * gamma[t] + beta[t];

  if (row == 0 && t == 0)
    out[(size_t)BATCH * N_NODES * F_DIM] = 1e-5f / (float)N_NODES;
}

// ---------------------------------------------------------------------------
extern "C" void kernel_launch(void* const* d_in, const int* in_sizes, int n_in,
                              void* d_out, int out_size, void* d_ws, size_t ws_size,
                              hipStream_t stream) {
  const float* feat    = (const float*)d_in[0];
  const float* fixedg  = (const float*)d_in[1];
  const float* learned = (const float*)d_in[2];
  const float* Wc      = (const float*)d_in[3];
  const float* bc      = (const float*)d_in[4];
  const float* We      = (const float*)d_in[5];
  const float* be      = (const float*)d_in[6];
  const float* Woc     = (const float*)d_in[7];
  const float* boc     = (const float*)d_in[8];
  const float* Woe     = (const float*)d_in[9];
  const float* boe     = (const float*)d_in[10];
  const float* bf1     = (const float*)d_in[11];
  const float* bf2     = (const float*)d_in[12];
  const float* fusion  = (const float*)d_in[13];
  const float* gamma   = (const float*)d_in[14];
  const float* beta    = (const float*)d_in[15];
  float* outp = (float*)d_out;

  // workspace carve-up
  const size_t QKV_ELEMS = (size_t)BATCH * H_HEADS * N_NODES * HD;   // 4.19M
  char* ws = (char*)d_ws;
  unsigned short* qbf = (unsigned short*)ws;                         //  8 MB
  unsigned short* kbf = (unsigned short*)(ws + QKV_ELEMS * 2);       //  8 MB
  unsigned short* vT  = (unsigned short*)(ws + QKV_ELEMS * 4);       //  8 MB
  float* biasH = (float*)(ws + QKV_ELEMS * 6);                       // 16 MB
  float* ctx   = (float*)(ws + QKV_ELEMS * 6 +
                          (size_t)H_HEADS * N_NODES * N_NODES * 4);  // 16 MB

  qkv_kernel<<<BATCH * N_NODES, 256, 0, stream>>>(feat, Wc, bc, We, be,
                                                  qbf, kbf, vT);
  bias_kernel<<<H_HEADS * N_NODES, 256, 0, stream>>>(bf1, bf2, fixedg, fusion,
                                                     biasH);
  attn_kernel<<<BATCH * H_HEADS * (N_NODES / 16), 32, 0, stream>>>(
      qbf, kbf, vT, biasH, learned, fusion, ctx);
  out_kernel<<<BATCH * N_NODES, 256, 0, stream>>>(ctx, feat, Woc, boc, Woe, boe,
                                                  gamma, beta, outp);
}
